// LSTMModel_27358941676368
// MI455X (gfx1250) — matrix operand order repaired and checked by reading, not compiled, not run on hardware
//
#include <hip/hip_runtime.h>
#include <cstdint>
#include <cstddef>

// ---------------------------------------------------------------------------
// 2-layer LSTM (B=256,T=512,D=64,H=512) + FC on last step, for gfx1250.
//  - V_WMMA_F32_16X16X32_BF16 for both GEMMs (fp32 accum, fp32 cell state)
//  - weight tiles staged to LDS via GLOBAL_LOAD_ASYNC_TO_LDS_B128 (ASYNCcnt),
//    SADDR+voffset form, double-buffered, shared by all 4 waves of the WG
//  - raw split barrier (s_barrier_signal/-1 + s_barrier_wait/-1) so the A-
//    fragment global prefetch is NOT drained at each chunk barrier
//  - wavefront schedule: one launch runs layer0 step k AND layer1 step k-1
//    (independent), halving the sequential launch chain (513 launches).
// Workspace budget: ~8.3 MB.
// ---------------------------------------------------------------------------

typedef __bf16          v16bf    __attribute__((ext_vector_type(16)));
typedef float           v8f      __attribute__((ext_vector_type(8)));
typedef unsigned short  ushort8  __attribute__((ext_vector_type(8)));
typedef unsigned short  ushort16 __attribute__((ext_vector_type(16)));

namespace {
constexpr int kB = 256;    // batch
constexpr int kT = 512;    // timesteps
constexpr int kD = 64;     // layer-0 input dim
constexpr int kH = 512;    // hidden
constexpr int kG = 4 * kH; // 2048 gate rows
constexpr int kWavesPerWG = 4;
}

__device__ __forceinline__ unsigned short f2bf_bits(float f) {
  __bf16 b = (__bf16)f;                       // native cvt on gfx1250
  union { __bf16 b; unsigned short u; } v; v.b = b;
  return v.u;
}
__device__ __forceinline__ float bf2f(unsigned short h) {
  union { unsigned u; float f; } v; v.u = ((unsigned)h) << 16; return v.f;
}
__device__ __forceinline__ float fast_sigmoid(float x) {
  return 1.0f / (1.0f + __expf(-x));
}
__device__ __forceinline__ float fast_tanh(float x) {
  float ax = fabsf(x);
  float e  = __expf(-2.0f * ax);
  float t  = (1.0f - e) / (1.0f + e);
  return copysignf(t, x);
}

// ---- gfx1250 async copy global->LDS, SADDR form, tracked by ASYNCcnt ------
// GVS addressing: mem = SADDR64 + VADDR32 + IOFFSET; the IOFFSET is added to
// BOTH the LDS and global addresses, so "offset:16" reuses the same operands.
__device__ __forceinline__ void async_copy_b128_pair(uint32_t lds_off, uint64_t sbase,
                                                     uint32_t voff) {
  asm volatile("global_load_async_to_lds_b128 %0, %1, %2"
               :: "v"(lds_off), "v"(voff), "s"(sbase) : "memory");
  asm volatile("global_load_async_to_lds_b128 %0, %1, %2 offset:16"
               :: "v"(lds_off), "v"(voff), "s"(sbase) : "memory");
}
__device__ __forceinline__ void wait_asynccnt0() {
  asm volatile("s_wait_asynccnt 0x0" ::: "memory");
}
// Plain workgroup split barrier WITHOUT fence semantics: LDS visibility is
// already guaranteed by s_wait_asynccnt 0 (per wave) + barrier ordering.
__device__ __forceinline__ void wg_barrier_raw() {
  asm volatile("s_barrier_signal -1\n\ts_barrier_wait -1" ::: "memory");
}

// -------------------- A-fragment builders (ISA 7.12.2, 16-bit A 16x32) -----
// Lane L: m = L%16, half = L/16.  Slot j (0..15):
//   j in [0,8)  -> K = kc + half*8 + j          (contiguous 16B)
//   j in [8,16) -> K = kc + 16 + half*8 + (j-8) (contiguous 16B)
__device__ __forceinline__ v16bf load_afrag_bf16(const unsigned short* __restrict__ base) {
  union { ushort8 u[2]; v16bf v; } a;
  a.u[0] = *(const ushort8*)(base);
  a.u[1] = *(const ushort8*)(base + 16);
  return a.v;
}
__device__ __forceinline__ v16bf load_afrag_f32(const float* __restrict__ base) {
  union { unsigned short u[16]; v16bf v; } a;
#pragma unroll
  for (int j = 0; j < 8; ++j) a.u[j]     = f2bf_bits(base[j]);
#pragma unroll
  for (int j = 0; j < 8; ++j) a.u[8 + j] = f2bf_bits(base[16 + j]);
  return a.v;
}

// -------------------- fused LSTM step body ---------------------------------
// One wave owns a 16(batch) x 16(h-unit) patch and accumulates all 4 gates.
// gates = x_t @ Wih^T + h_prev @ Whh^T + b; weight tiles come through LDS.
template<int DK, bool XF32>
__device__ __forceinline__ void lstm_step_body(
    const void* __restrict__ xin_, int ldx,
    const unsigned short* __restrict__ w_ih,   // [4H, DK] bf16
    const unsigned short* __restrict__ w_hh,   // [4H, H]  bf16
    const float* __restrict__ bias,            // [4H] = b_ih + b_hh
    const unsigned short* __restrict__ h_prev, // [B, H] bf16
    float* __restrict__ c,                     // [B, H] fp32, in/out
    unsigned short* __restrict__ h_out,        // [B, H] bf16
    unsigned short (* __restrict__ smem)[4 * 16 * 32])
{
  const int lane = threadIdx.x & 31;
  const int wave = threadIdx.x >> 5;
  const int mi   = lane & 15;
  const int half = lane >> 4;
  const int m0 = (blockIdx.x * kWavesPerWG + wave) * 16;
  const int n0 = blockIdx.y * 16;

  // Cooperative async stage of one 4KB tile: 128 threads x 32B.
  // LDS row r = g*16+n holds W[g*H + n0 + n, kc .. kc+31] (k contiguous), so a
  // lane's B-fragment (n=mi, K=half*8..+7 and 16+half*8..+7) is two 16B pieces.
  const int r    = threadIdx.x >> 1;          // 0..63 : g*16 + n
  const int part = (threadIdx.x & 1) * 16;    // ushort offset within row
  const int sg   = r >> 4;                    // gate
  const int sn   = r & 15;                    // n within tile
  auto stage = [&](const unsigned short* __restrict__ W, int ldw, int kc, int p) {
    const uint32_t voff = (uint32_t)(((sg * kH + n0 + sn) * ldw + kc + part) * 2);
    const uint32_t dst  = (uint32_t)(uintptr_t)&smem[p][r * 32 + part];
    async_copy_b128_pair(dst, (uint64_t)(uintptr_t)W, voff);
  };

  constexpr int NC0 = DK / 32;        // input-projection chunks
  constexpr int NCT = NC0 + kH / 32;  // + recurrent chunks

  v8f acc[4];
#pragma unroll
  for (int g = 0; g < 4; ++g) acc[g] = (v8f){0.f,0.f,0.f,0.f,0.f,0.f,0.f,0.f};

  stage(w_ih, DK, 0, 0);              // prologue
#pragma unroll
  for (int ci = 0; ci < NCT; ++ci) {
    const int p = ci & 1;

    // A fragment first (longest latency, independent of LDS; survives the
    // raw barrier below without being drained)
    v16bf a;
    if (ci < NC0) {
      const int kc = ci * 32;
      if (XF32) a = load_afrag_f32((const float*)xin_ +
                                   (size_t)(m0 + mi) * ldx + kc + half * 8);
      else      a = load_afrag_bf16((const unsigned short*)xin_ +
                                    (size_t)(m0 + mi) * ldx + kc + half * 8);
    } else {
      const int kc = (ci - NC0) * 32;
      a = load_afrag_bf16(h_prev + (size_t)(m0 + mi) * kH + kc + half * 8);
    }

    wait_asynccnt0();                 // this wave's async writes to smem[p] done
    wg_barrier_raw();                 // all waves' writes done (no vmem drain)
    if (ci + 1 < NCT) {
      if (ci + 1 < NC0) stage(w_ih, DK, (ci + 1) * 32, p ^ 1);
      else              stage(w_hh, kH, (ci + 1 - NC0) * 32, p ^ 1);
    }

    // Pre-load all 4 B-fragments (8x ds_load_b128, staggered waits), then WMMA
    union { ushort16 u; v16bf v; } bfr[4];
#pragma unroll
    for (int g = 0; g < 4; ++g) {
      const unsigned short* q = &smem[p][(g * 16 + mi) * 32 + half * 8];
      bfr[g].u = __builtin_shufflevector(*(const ushort8*)q, *(const ushort8*)(q + 16),
                                         0, 1, 2, 3, 4, 5, 6, 7, 8, 9, 10, 11, 12, 13, 14, 15);
    }
#pragma unroll
    for (int g = 0; g < 4; ++g)
      acc[g] = __builtin_amdgcn_wmma_f32_16x16x32_bf16(
          false, a, false, bfr[g].v, (short)0, acc[g], false, false);
  }

  // ---- gate nonlinearities + cell/hidden update ----
  // C/D layout: lane L holds n = L%16; VGPR v holds row m = v + 8*(L/16).
  const float bi  = bias[0 * kH + n0 + mi];
  const float bff = bias[1 * kH + n0 + mi];
  const float bg  = bias[2 * kH + n0 + mi];
  const float bo  = bias[3 * kH + n0 + mi];
#pragma unroll
  for (int v = 0; v < 8; ++v) {
    const int m = m0 + v + 8 * half;
    const size_t idx = (size_t)m * kH + n0 + mi;
    const float ig = fast_sigmoid(acc[0][v] + bi);
    const float fg = fast_sigmoid(acc[1][v] + bff);
    const float gg = fast_tanh(acc[2][v] + bg);
    const float og = fast_sigmoid(acc[3][v] + bo);
    const float cn = fg * c[idx] + ig * gg;
    c[idx] = cn;
    h_out[idx] = f2bf_bits(og * fast_tanh(cn));
  }
}

// -------------------- dual-layer wavefront step ----------------------------
// blockIdx.z == 0 : layer0 step k      (skipped when k == kT)
// blockIdx.z == 1 : layer1 step k-1    (skipped when k == 0)
__global__ __launch_bounds__(kWavesPerWG * 32) void k_lstm_dual_step(
    int k,
    const float* __restrict__ x,
    const unsigned short* __restrict__ wih0, const unsigned short* __restrict__ whh0,
    const float* __restrict__ bias0,
    const unsigned short* __restrict__ wih1, const unsigned short* __restrict__ whh1,
    const float* __restrict__ bias1,
    unsigned short* __restrict__ h0a, unsigned short* __restrict__ h0b,
    unsigned short* __restrict__ h1a, unsigned short* __restrict__ h1b,
    float* __restrict__ c0, float* __restrict__ c1)
{
  __shared__ __align__(32) unsigned short smem[2][4 * 16 * 32];  // 8 KB

  if (blockIdx.z == 0) {
    if (k >= kT) return;
    unsigned short* h0prev = (k & 1) ? h0b : h0a;
    unsigned short* h0out  = (k & 1) ? h0a : h0b;
    lstm_step_body<kD, true>(x + (size_t)k * kD, kT * kD,
                             wih0, whh0, bias0, h0prev, c0, h0out, smem);
  } else {
    if (k == 0) return;
    const int t = k - 1;
    const unsigned short* h0cur = ((t + 1) & 1) ? h0b : h0a;  // layer0 out @ t
    unsigned short* h1prev = (t & 1) ? h1b : h1a;
    unsigned short* h1out  = ((t + 1) & 1) ? h1b : h1a;
    lstm_step_body<kH, false>(h0cur, kH,
                              wih1, whh1, bias1, h1prev, c1, h1out, smem);
  }
}

// -------------------- prep / epilogue kernels ------------------------------
__global__ void k_cvt_bf16(const float* __restrict__ s, unsigned short* __restrict__ d, int n) {
  int i = blockIdx.x * blockDim.x + threadIdx.x;
  if (i < n) d[i] = f2bf_bits(s[i]);
}
__global__ void k_vadd(const float* __restrict__ a, const float* __restrict__ b,
                       float* __restrict__ o, int n) {
  int i = blockIdx.x * blockDim.x + threadIdx.x;
  if (i < n) o[i] = a[i] + b[i];
}
__global__ void k_zero32(uint32_t* __restrict__ p, int n) {
  int i = blockIdx.x * blockDim.x + threadIdx.x;
  if (i < n) p[i] = 0u;
}
__global__ void k_fc(const unsigned short* __restrict__ h1,
                     const float* __restrict__ fc_w, const float* __restrict__ fc_b,
                     float* __restrict__ out) {
  int b = blockIdx.x * blockDim.x + threadIdx.x;
  if (b >= kB) return;
  float s = 0.f;
  for (int h = 0; h < kH; ++h) s += bf2f(h1[(size_t)b * kH + h]) * fc_w[h];
  out[b] = s + fc_b[0];
}

// -------------------- launcher ---------------------------------------------
extern "C" void kernel_launch(void* const* d_in, const int* in_sizes, int n_in,
                              void* d_out, int out_size, void* d_ws, size_t ws_size,
                              hipStream_t stream) {
  (void)in_sizes; (void)n_in; (void)out_size; (void)ws_size;
  const float* x      = (const float*)d_in[0];
  const float* w_ih0f = (const float*)d_in[1];
  const float* w_hh0f = (const float*)d_in[2];
  const float* b_ih0  = (const float*)d_in[3];
  const float* b_hh0  = (const float*)d_in[4];
  const float* w_ih1f = (const float*)d_in[5];
  const float* w_hh1f = (const float*)d_in[6];
  const float* b_ih1  = (const float*)d_in[7];
  const float* b_hh1  = (const float*)d_in[8];
  const float* fc_w   = (const float*)d_in[9];
  const float* fc_b   = (const float*)d_in[10];

  // ---- carve workspace (~8.3 MB) ----
  char* base = (char*)d_ws;
  size_t off = 0;
  auto carve = [&](size_t bytes) -> void* {
    off = (off + 255) & ~(size_t)255;
    void* r = base + off;
    off += bytes;
    return r;
  };
  unsigned short* wih0 = (unsigned short*)carve((size_t)kG * kD * 2);
  unsigned short* whh0 = (unsigned short*)carve((size_t)kG * kH * 2);
  unsigned short* wih1 = (unsigned short*)carve((size_t)kG * kH * 2);
  unsigned short* whh1 = (unsigned short*)carve((size_t)kG * kH * 2);
  float* bias0 = (float*)carve((size_t)kG * 4);
  float* bias1 = (float*)carve((size_t)kG * 4);
  // contiguous state block so one zero-kernel resets everything per launch
  const size_t SN = (size_t)kB * kH;                       // 131072 elements
  const size_t stateBytes = SN * (4 + 4 + 2 + 2 + 2 + 2);  // c0,c1 f32; h0a/b,h1a/b bf16
  char* state = (char*)carve(stateBytes);
  float* c0 = (float*)state;
  float* c1 = c0 + SN;
  unsigned short* h0a = (unsigned short*)(c1 + SN);
  unsigned short* h0b = h0a + SN;
  unsigned short* h1a = h0b + SN;
  unsigned short* h1b = h1a + SN;

  // ---- prep: weight conversion, bias fusion, state zeroing ----
  const int TPB = 256;
  k_cvt_bf16<<<(kG * kD + TPB - 1) / TPB, TPB, 0, stream>>>(w_ih0f, wih0, kG * kD);
  k_cvt_bf16<<<(kG * kH + TPB - 1) / TPB, TPB, 0, stream>>>(w_hh0f, whh0, kG * kH);
  k_cvt_bf16<<<(kG * kH + TPB - 1) / TPB, TPB, 0, stream>>>(w_ih1f, wih1, kG * kH);
  k_cvt_bf16<<<(kG * kH + TPB - 1) / TPB, TPB, 0, stream>>>(w_hh1f, whh1, kG * kH);
  k_vadd<<<(kG + TPB - 1) / TPB, TPB, 0, stream>>>(b_ih0, b_hh0, bias0, kG);
  k_vadd<<<(kG + TPB - 1) / TPB, TPB, 0, stream>>>(b_ih1, b_hh1, bias1, kG);
  const int stateWords = (int)(stateBytes / 4);
  k_zero32<<<(stateWords + TPB - 1) / TPB, TPB, 0, stream>>>((uint32_t*)state, stateWords);

  // ---- wavefront recurrence: launch k runs layer0@k and layer1@(k-1) ----
  dim3 sgrid(kB / (16 * kWavesPerWG), kH / 16, 2);  // (4, 32, 2)
  dim3 sblock(kWavesPerWG * 32);                    // 128 threads = 4 waves
  for (int k = 0; k <= kT; ++k) {
    k_lstm_dual_step<<<sgrid, sblock, 0, stream>>>(
        k, x, wih0, whh0, bias0, wih1, whh1, bias1,
        h0a, h0b, h1a, h1b, c0, c1);
  }

  // ---- FC on last hidden state (layer1 output of t=T-1 lands in h1a) ----
  k_fc<<<1, kB, 0, stream>>>(h1a, fc_w, fc_b, (float*)d_out);
}
